// LatentFactorPooling_46901042872713
// MI455X (gfx1250) — matrix operand order repaired and checked by reading, not compiled, not run on hardware
//
#include <hip/hip_runtime.h>

// ---------------------------------------------------------------------------
// LatentFactorPooling for MI455X (gfx1250, wave32, WMMA)
//   x:            (64, 256, 64, 44) f32
//   latent_basis: (4, 16, 256) f32
//   out: tokens (64, 256, 64) f32  ++  pres (64, 64) f32
// ---------------------------------------------------------------------------

typedef __attribute__((ext_vector_type(16))) __bf16 v16bf;
typedef __attribute__((ext_vector_type(8)))  __bf16 v8bf;
typedef __attribute__((ext_vector_type(8)))  float  v8f;
typedef __attribute__((ext_vector_type(4)))  float  v4f;

#define NB    64
#define CCH   256
#define HH    64
#define WW    44
#define GS    4
#define KL    16
#define HS    16
#define PP    704          // HS*WW pixels per stripe
#define NG    256          // NB*GS stripes
#define HW    (HH*WW)      // 2816
#define TOPK  88           // int(704*0.125)
#define TEMP_INV 8.0f
#define GATE  0.05f
#define EPSV  1e-6f

static __device__ __forceinline__ v16bf cat8(v8bf lo, v8bf hi) {
    return __builtin_shufflevector(lo, hi, 0, 1, 2, 3, 4, 5, 6, 7,
                                           8, 9, 10, 11, 12, 13, 14, 15);
}

// ---------------------------------------------------------------------------
// Pass 0: L2-normalize latent basis, bf16, layout [g][k][c] (row = g*16+k).
// B operand then needs only 16 *contiguous* bf16 per lane.
// ---------------------------------------------------------------------------
__global__ void __launch_bounds__(256)
k_basis(const float* __restrict__ lb, __bf16* __restrict__ lbn) {
    int row = blockIdx.x;                 // 0..63
    int c = threadIdx.x;                  // 0..255
    float v = lb[(size_t)row * CCH + c];
    __shared__ float red[256];
    red[c] = v * v;
    __syncthreads();
    for (int s = 128; s > 0; s >>= 1) {
        if (c < s) red[c] += red[c + s];
        __syncthreads();
    }
    float inv = 1.0f / fmaxf(sqrtf(red[0]), 1e-12f);
    lbn[(size_t)row * CCH + c] = (__bf16)(v * inv);
}

// ---------------------------------------------------------------------------
// Pass 1 (fused): per-stripe avg-pool + energy mask + routing GEMM (WMMA)
//                 + softmax + masking -> suppT[k][p], colsum[k]
// One block per stripe, 704 threads = 22 waves, 1 pixel/thread,
// 2 pixel-tiles per wave.  Channel loop: 8 tiles of 32.
// poolT is pixel-major (64 B/row) so each WMMA A operand is two b128 DS loads.
// ---------------------------------------------------------------------------
__global__ void __launch_bounds__(704)
k_route(const float* __restrict__ x, const __bf16* __restrict__ lbn,
        float* __restrict__ suppT, float* __restrict__ colsum) {
    extern __shared__ char smem[];
    float*  sfT = (float*)smem;            // 32*704 f32  raw tile
    float*  npS = sfT + 32 * PP;           // 704         ||pool||^2
    float*  amS = npS + PP;                // 704         active mask
    float*  csS = amS + PP;                // 16          colsum
    float*  red = csS + 16;                // 32          wave partials
    __bf16* poolT = (__bf16*)(red + 32);   // 704 x 32 bf16, pixel-major

    int s = blockIdx.x;
    int nIdx = s / GS, g = s % GS;
    const float* base = x + (size_t)nIdx * CCH * HW + (size_t)g * HS * WW;

    int t = threadIdx.x, lane = t & 31, wv = t >> 5;   // 22 waves
    int nk = lane & 15;
    int r0 = t / WW, w0 = t - r0 * WW;                 // this thread's pixel

    if (t < 16) csS[t] = 0.f;

    float np0 = 0.f, e0 = 0.f;
    v8f zero8 = {0, 0, 0, 0, 0, 0, 0, 0};
    v8f acc0 = zero8, acc1 = zero8;
    const __bf16* lrow = lbn + (size_t)(g * KL + nk) * CCH + ((lane < 16) ? 0 : 16);
    int abase = (lane < 16) ? 0 : 16;

#pragma unroll 1
    for (int ct = 0; ct < 8; ++ct) {
        __syncthreads();
        // coalesced tile load (c = i, p = t exactly); fuse energy accumulation
#pragma unroll
        for (int c = 0; c < 32; ++c) {
            float v = base[(size_t)(ct * 32 + c) * HW + t];
            sfT[c * PP + t] = v;
            e0 += v * v;
        }
        __syncthreads();

        // 3x3 avg pool (count_include_pad -> /9 always) + pool-norm
        union { v8bf v[4]; __bf16 e[32]; } prow;
#pragma unroll
        for (int c = 0; c < 32; ++c) {
            const float* pl = sfT + c * PP;
            float s0 = 0.f;
#pragma unroll
            for (int dr = -1; dr <= 1; ++dr) {
                int rr = r0 + dr;
                if ((unsigned)rr < HS) {
                    const float* rp = pl + rr * WW;
                    if (w0 > 0)      s0 += rp[w0 - 1];
                    s0 += rp[w0];
                    if (w0 < WW - 1) s0 += rp[w0 + 1];
                }
            }
            s0 *= (1.f / 9.f);
            prow.e[c] = (__bf16)s0;
            np0 += s0 * s0;
        }
        {
            v8bf* dst = (v8bf*)(poolT + (size_t)t * 32);
            dst[0] = prow.v[0]; dst[1] = prow.v[1];
            dst[2] = prow.v[2]; dst[3] = prow.v[3];
        }
        __syncthreads();

        // B tile: 16 contiguous bf16 per lane (lanes 0-15: K=0..15; 16-31: K=16..31)
        v16bf bt = cat8(*(const v8bf*)(lrow + ct * 32),
                        *(const v8bf*)(lrow + ct * 32 + 8));
        // A tiles: two b128 DS loads per lane per tile (contiguous k = 16h + s, same as B)
#pragma unroll
        for (int q = 0; q < 2; ++q) {
            int prowi = (wv * 2 + q) * 16 + nk;
            const __bf16* ap = poolT + (size_t)prowi * 32 + abase;
            v16bf at = cat8(*(const v8bf*)ap, *(const v8bf*)(ap + 8));
            if (q == 0)
                acc0 = __builtin_amdgcn_wmma_f32_16x16x32_bf16(
                    false, at, false, bt, (short)0, acc0, false, false);
            else
                acc1 = __builtin_amdgcn_wmma_f32_16x16x32_bf16(
                    false, at, false, bt, (short)0, acc1, false, false);
        }
    }

    // ---- energy mask (mean over c, normalized by stripe max, gate+fallback)
    e0 *= (1.f / CCH);
    npS[t] = np0;

    float m = e0;
    for (int d = 16; d >= 1; d >>= 1) m = fmaxf(m, __shfl_xor(m, d, 32));
    if (lane == 0) red[wv] = m;
    __syncthreads();
    if (t == 0) {
        float mm = red[0];
        for (int i = 1; i < 22; ++i) mm = fmaxf(mm, red[i]);
        red[24] = mm;
    }
    __syncthreads();
    float inv = 1.f / fmaxf(red[24], EPSV);
    float a0 = (e0 * inv > GATE) ? 1.f : 0.f;
    float cnt = a0;
    for (int d = 16; d >= 1; d >>= 1) cnt += __shfl_xor(cnt, d, 32);
    __syncthreads();
    if (lane == 0) red[wv] = cnt;
    __syncthreads();
    if (t == 0) {
        float cs = 0.f;
        for (int i = 0; i < 22; ++i) cs += red[i];
        red[25] = (cs <= 0.f) ? 1.f : 0.f;
    }
    __syncthreads();
    bool fb = red[25] > 0.5f;
    amS[t] = fb ? (e0 > 0.f ? 1.f : 0.f) : a0;
    __syncthreads();

    // ---- logits = acc/(||pool||*TEMP); softmax over 16 latents (16-lane
    // group holds all k of one pixel -> shfl_xor 1..8); mask; store to
    // suppT[s][k][p] with vector stores.
    float cpart = 0.f;
    int mrowoff = (lane < 16) ? 0 : 8;
    float* sp = suppT + ((size_t)s * KL + nk) * PP;
#pragma unroll
    for (int q = 0; q < 2; ++q) {
        int tIdx = wv * 2 + q;
        union { v4f v[2]; float e[8]; } out8;
#pragma unroll
        for (int j = 0; j < 8; ++j) {
            int p = tIdx * 16 + j + mrowoff;
            float av = (q == 0) ? acc0[j] : acc1[j];
            float nrm = fmaxf(sqrtf(npS[p]), 1e-12f);
            float logit = av / nrm * TEMP_INV;
            float mx = logit;
            for (int d = 8; d >= 1; d >>= 1) mx = fmaxf(mx, __shfl_xor(mx, d, 32));
            float ex = __expf(logit - mx);
            float sum = ex;
            for (int d = 8; d >= 1; d >>= 1) sum += __shfl_xor(sum, d, 32);
            float sv = (ex / sum) * amS[p];
            out8.e[j] = sv;
            cpart += sv;
        }
        v4f* dst = (v4f*)(sp + tIdx * 16 + mrowoff);
        dst[0] = out8.v[0];
        dst[1] = out8.v[1];
    }
    atomicAdd(&csS[nk], cpart);
    __syncthreads();
    if (t < 16) colsum[(size_t)s * KL + t] = csS[t];
}

// ---------------------------------------------------------------------------
// Pass 2: tokens[k,c] = sum_p (suppT[k,p]/colsum[k]) * sf[c,p]  (WMMA)
// One block per stripe, 8 waves; wave w owns channel tiles w and w+8.
// x re-read hits L2 (184 MB < 192 MB).  All operands are v4f vector loads.
// ---------------------------------------------------------------------------
__global__ void __launch_bounds__(256)
k_tokens(const float* __restrict__ x, const float* __restrict__ suppT,
         const float* __restrict__ colsum, float* __restrict__ tokens) {
    int s = blockIdx.x;
    int nIdx = s / GS, g = s % GS;
    const float* base = x + (size_t)nIdx * CCH * HW + (size_t)g * HS * WW;
    int t = threadIdx.x, lane = t & 31, wv = t >> 5;
    int nk = lane & 15;
    float rcs = 1.0f / fmaxf(colsum[(size_t)s * KL + nk], EPSV);

    const float* ap = suppT + ((size_t)s * KL + nk) * PP;   // row k = nk
    int abase = (lane < 16) ? 0 : 16;
    int boff = (lane < 16) ? 0 : 16;
    const float* b0p = base + (size_t)(wv * 16 + nk) * HW + boff;
    const float* b1p = base + (size_t)((wv + 8) * 16 + nk) * HW + boff;

    v8f zero8 = {0, 0, 0, 0, 0, 0, 0, 0};
    v8f acc0 = zero8, acc1 = zero8;

#pragma unroll 2
    for (int pt = 0; pt < 22; ++pt) {               // 22*32 = 704 = K loop
        // A = pw^T : 16 latents x 32 pixels (A layout -> 4 contiguous v4f)
        const float* a = ap + pt * 32 + abase;
        v4f a0 = *(const v4f*)(a);
        v4f a1 = *(const v4f*)(a + 4);
        v4f a2 = *(const v4f*)(a + 8);
        v4f a3 = *(const v4f*)(a + 12);
        v16bf at;
#pragma unroll
        for (int i = 0; i < 4; ++i) {
            at[i]      = (__bf16)(a0[i] * rcs);
            at[4 + i]  = (__bf16)(a1[i] * rcs);
            at[8 + i]  = (__bf16)(a2[i] * rcs);
            at[12 + i] = (__bf16)(a3[i] * rcs);
        }
        // B = sf : 32 pixels x 16 channels; 16 contiguous floats per lane
        v16bf bt0, bt1;
#pragma unroll
        for (int m = 0; m < 4; ++m) {
            v4f c0 = *(const v4f*)(b0p + pt * 32 + 4 * m);
            v4f c1 = *(const v4f*)(b1p + pt * 32 + 4 * m);
#pragma unroll
            for (int i = 0; i < 4; ++i) {
                bt0[4 * m + i] = (__bf16)c0[i];
                bt1[4 * m + i] = (__bf16)c1[i];
            }
        }
        acc0 = __builtin_amdgcn_wmma_f32_16x16x32_bf16(
            false, at, false, bt0, (short)0, acc0, false, false);
        acc1 = __builtin_amdgcn_wmma_f32_16x16x32_bf16(
            false, at, false, bt1, (short)0, acc1, false, false);
    }
    // D layout: VGPR j -> M=j (lanes 0-15) / M=j+8 (lanes 16-31), N = lane%16
    int mo = (lane < 16) ? 0 : 8;
    int c0 = wv * 16 + nk, c1 = (wv + 8) * 16 + nk;
#pragma unroll
    for (int j = 0; j < 8; ++j) {
        int k = j + mo;
        tokens[((size_t)nIdx * CCH + c0) * (GS * KL) + g * KL + k] = acc0[j];
        tokens[((size_t)nIdx * CCH + c1) * (GS * KL) + g * KL + k] = acc1[j];
    }
}

// ---------------------------------------------------------------------------
// Pass 3: presence = mean of top-88 of each suppT row (704 values).
// One wave per (stripe, k) row; suppT[k][p] layout -> loads fully coalesced.
// 88th-largest via 28-step threshold bisection with wave reductions.
// ---------------------------------------------------------------------------
__global__ void __launch_bounds__(256)
k_presence(const float* __restrict__ suppT, float* __restrict__ presun) {
    int row = blockIdx.x * 8 + (threadIdx.x >> 5);   // 0..4095
    int lane = threadIdx.x & 31;
    int s = row >> 4, k = row & 15;
    const float* sp = suppT + ((size_t)s * KL + k) * PP;

    float v[22];
    float mx = 0.f;
#pragma unroll
    for (int i = 0; i < 22; ++i) {
        v[i] = sp[lane + 32 * i];
        mx = fmaxf(mx, v[i]);
    }
    for (int d = 16; d >= 1; d >>= 1) mx = fmaxf(mx, __shfl_xor(mx, d, 32));

    float lo = 0.f, hi = mx;
    for (int it = 0; it < 28; ++it) {
        float mid = 0.5f * (lo + hi);
        int c = 0;
#pragma unroll
        for (int i = 0; i < 22; ++i) c += (v[i] > mid) ? 1 : 0;
        for (int d = 16; d >= 1; d >>= 1) c += __shfl_xor(c, d, 32);
        if (c >= TOPK) lo = mid; else hi = mid;
    }
    float thr = lo, sum = 0.f;
    int c = 0;
#pragma unroll
    for (int i = 0; i < 22; ++i)
        if (v[i] > thr) { sum += v[i]; ++c; }
    for (int d = 16; d >= 1; d >>= 1) {
        sum += __shfl_xor(sum, d, 32);
        c   += __shfl_xor(c, d, 32);
    }
    float mean = (sum + (float)(TOPK - c) * thr) * (1.0f / TOPK);
    if (lane == 0) presun[row] = mean;
}

// Pass 4: normalize presence per image (64 values per n).
__global__ void __launch_bounds__(64)
k_presnorm(const float* __restrict__ presun, float* __restrict__ out) {
    int nIdx = blockIdx.x, t = threadIdx.x;
    __shared__ float red[64];
    float v = presun[(size_t)nIdx * 64 + t];
    red[t] = v;
    __syncthreads();
    for (int w = 32; w > 0; w >>= 1) {
        if (t < w) red[t] += red[t + w];
        __syncthreads();
    }
    out[(size_t)nIdx * 64 + t] = v / fmaxf(red[0], EPSV);
}

// ---------------------------------------------------------------------------
extern "C" void kernel_launch(void* const* d_in, const int* in_sizes, int n_in,
                              void* d_out, int out_size, void* d_ws, size_t ws_size,
                              hipStream_t stream) {
    const float* x  = (const float*)d_in[0];
    const float* lb = (const float*)d_in[1];
    float* out = (float*)d_out;

    char* ws = (char*)d_ws;
    __bf16* lbn    = (__bf16*)ws;                              //    32,768 B
    float*  suppT  = (float*)(ws + 32768);                     // 11,534,336 B
    float*  colsum = (float*)(ws + 32768 + 11534336);          //    16,384 B
    float*  presun = (float*)(ws + 32768 + 11534336 + 16384);  //    16,384 B

    k_basis<<<GS * KL, 256, 0, stream>>>(lb, lbn);

    size_t shmem = (size_t)(32 * PP + PP + PP + 16 + 32) * sizeof(float)
                 + (size_t)(32 * PP) * sizeof(__bf16);         // ~141 KB (WGP: 320 KB)
    k_route<<<NG, 704, shmem, stream>>>(x, lbn, suppT, colsum);

    k_tokens<<<NG, 256, 0, stream>>>(x, suppT, colsum, out);

    k_presence<<<NG * KL / 8, 256, 0, stream>>>(suppT, presun);

    k_presnorm<<<NB, 64, 0, stream>>>(presun, out + (size_t)NB * CCH * GS * KL);
}
